// MultiHeadAttentionQuantum_65481071409130
// MI455X (gfx1250) — matrix operand order repaired and checked by reading, not compiled, not run on hardware
//
#include <hip/hip_runtime.h>
#include <hip/hip_bf16.h>

typedef __attribute__((ext_vector_type(16))) _Float16 v16h;
typedef __attribute__((ext_vector_type(8)))  float    v8f;

#define B_  4
#define S_  2048
#define H_  16
#define DK_ 8
#define E_  128   // H_*DK_

// ---------------------------------------------------------------------------
// Kernel 1: quantum features  q[b,h,s,i] = cumprod_i cos(x[b,s,h*8+i] + theta[i])
// Stored f16, layout [b][h][s][8] (16B rows) so attention tiles load as b128.
// ---------------------------------------------------------------------------
__global__ void qfeat_kernel(const float* __restrict__ x,
                             const float* __restrict__ theta,
                             _Float16* __restrict__ qf) {
  int idx = blockIdx.x * blockDim.x + threadIdx.x;       // one thread per (b,s,h)
  if (idx >= B_ * S_ * H_) return;
  int h = idx % H_;
  int s = (idx / H_) % S_;
  int b = idx / (H_ * S_);
  const float* xp = x + ((size_t)(b * S_ + s)) * E_ + h * DK_;
  _Float16* qp = qf + (((size_t)(b * H_ + h)) * S_ + s) * DK_;
  float prod = 1.0f;
#pragma unroll
  for (int i = 0; i < DK_; ++i) {
    prod *= __cosf(xp[i] + theta[i]);
    qp[i] = (_Float16)prod;
  }
}

// ---------------------------------------------------------------------------
// Kernel 2: flash attention. One wave (32 threads) owns 32 query rows of one
// (b,h): two 16-row A fragments sharing each staged K/V tile.
// Per 32-key tile: one coalesced b128 load/lane stages 32x8 f16 into LDS,
// then 4 score WMMAs + 2 PV WMMAs (v_wmma_f32_16x16x32_f16), online softmax
// via width-16 xor-shuffles (matches wave32 C-fragment row layout).
// launch_bounds(32,1): single-wave block, full VGPR budget -> no spills;
// tile-0 softmax is scheduled between the two score-WMMA pairs to halve the
// peak number of live score fragments.
// ---------------------------------------------------------------------------
__global__ void __launch_bounds__(32, 1)
attn_kernel(const _Float16* __restrict__ qf,
            float* __restrict__ attn) {
  __shared__ _Float16 lds_kv[32][8];    // staged K/V tile  [key][d]
  __shared__ _Float16 lds_p[32][32];    // P tiles          [query m][key k]

  const int lane = threadIdx.x;                 // 0..31
  const int bid  = blockIdx.x;
  const int qt = bid % (S_ / 32);
  const int h  = (bid / (S_ / 32)) % H_;
  const int b  = bid / ((S_ / 32) * H_);
  const _Float16* kf = qf + ((size_t)(b * H_ + h)) * S_ * DK_;
  const int q0 = qt * 32;
  const float inv_sqrt_d = 0.35355339059327373f;   // 1/sqrt(8)
  const int ncol = lane & 15;
  const int gsel = (lane < 16) ? 0 : 8;            // C-fragment row offset

  // --- Q as two A-fragments (M=16, K=8 valid), pre-scaled by 1/sqrt(d).
  v16h aq0 = {}, aq1 = {};
  if (lane < 16) {
    const _Float16* qp0 = kf + (size_t)(q0 + lane) * DK_;
    const _Float16* qp1 = kf + (size_t)(q0 + 16 + lane) * DK_;
#pragma unroll
    for (int i = 0; i < DK_; ++i) {
      aq0[i] = (_Float16)((float)qp0[i] * inv_sqrt_d);
      aq1[i] = (_Float16)((float)qp1[i] * inv_sqrt_d);
    }
  }

  float mrow0[8], lrow0[8], mrow1[8], lrow1[8];
#pragma unroll
  for (int r = 0; r < 8; ++r) {
    mrow0[r] = -1e30f; lrow0[r] = 0.0f;
    mrow1[r] = -1e30f; lrow1[r] = 0.0f;
  }
  v8f o0 = {}, o1 = {};

  for (int kt = 0; kt < S_; kt += 32) {
    // --- stage K/V tile: lane loads key row (kt+lane), 16B coalesced.
    *reinterpret_cast<float4*>(&lds_kv[lane][0]) =
        *reinterpret_cast<const float4*>(kf + (size_t)(kt + lane) * DK_);
    __syncthreads();

    // --- K tiles as B-fragments from LDS (N=key column, K=d=8 valid rows).
    v16h bk0 = {}, bk1 = {};
    if (lane < 16) {
#pragma unroll
      for (int i = 0; i < DK_; ++i) {
        bk0[i] = lds_kv[lane][i];
        bk1[i] = lds_kv[16 + lane][i];
      }
    }
    v8f zero = {};

    // ===== query sub-tile 0: scores, then softmax (frees s00/s01 early) =====
    {
      v8f s00 = __builtin_amdgcn_wmma_f32_16x16x32_f16(false, aq0, false, bk0,
                                                       (short)0, zero, false, false);
      v8f s01 = __builtin_amdgcn_wmma_f32_16x16x32_f16(false, aq0, false, bk1,
                                                       (short)0, zero, false, false);
#pragma unroll
      for (int r = 0; r < 8; ++r) {
        float v0 = s00[r], v1 = s01[r];
        float tm = fmaxf(v0, v1);
#pragma unroll
        for (int off = 8; off >= 1; off >>= 1)
          tm = fmaxf(tm, __shfl_xor(tm, off, 16));
        float mnew = fmaxf(mrow0[r], tm);
        float p0 = __expf(v0 - mnew);
        float p1 = __expf(v1 - mnew);
        float ts = p0 + p1;
#pragma unroll
        for (int off = 8; off >= 1; off >>= 1)
          ts += __shfl_xor(ts, off, 16);
        float corr = __expf(mrow0[r] - mnew);
        lrow0[r] = lrow0[r] * corr + ts;
        o0[r] *= corr;
        mrow0[r] = mnew;
        int m = r + gsel;
        lds_p[m][ncol]      = (_Float16)p0;
        lds_p[m][16 + ncol] = (_Float16)p1;
      }
    }

    // ===== query sub-tile 1 =====
    {
      v8f s10 = __builtin_amdgcn_wmma_f32_16x16x32_f16(false, aq1, false, bk0,
                                                       (short)0, zero, false, false);
      v8f s11 = __builtin_amdgcn_wmma_f32_16x16x32_f16(false, aq1, false, bk1,
                                                       (short)0, zero, false, false);
#pragma unroll
      for (int r = 0; r < 8; ++r) {
        float v0 = s10[r], v1 = s11[r];
        float tm = fmaxf(v0, v1);
#pragma unroll
        for (int off = 8; off >= 1; off >>= 1)
          tm = fmaxf(tm, __shfl_xor(tm, off, 16));
        float mnew = fmaxf(mrow1[r], tm);
        float p0 = __expf(v0 - mnew);
        float p1 = __expf(v1 - mnew);
        float ts = p0 + p1;
#pragma unroll
        for (int off = 8; off >= 1; off >>= 1)
          ts += __shfl_xor(ts, off, 16);
        float corr = __expf(mrow1[r] - mnew);
        lrow1[r] = lrow1[r] * corr + ts;
        o1[r] *= corr;
        mrow1[r] = mnew;
        int m = r + gsel;
        lds_p[16 + m][ncol]      = (_Float16)p0;
        lds_p[16 + m][16 + ncol] = (_Float16)p1;
      }
    }
    __syncthreads();

    // --- P -> A-fragments from LDS: lane holds row M=lane&15, K pattern:
    //     lanes<16 get K {0..7,16..23}, lanes>=16 get K {8..15,24..31}.
    v16h ap0, ap1;
    {
      const int mr = lane & 15;
      const int kb = (lane < 16) ? 0 : 8;
#pragma unroll
      for (int hh = 0; hh < 16; ++hh) {
        int k = (hh < 8) ? (kb + hh) : (kb + 8 + hh);
        ap0[hh] = lds_p[mr][k];
        ap1[hh] = lds_p[16 + mr][k];
      }
    }
    // --- V as B-fragment from LDS: column N = d-index (valid <8),
    //     K = key-in-tile; lanes 0-15 hold K {0..7,16..23}, lanes 16-31 K {8..15,24..31}.
    v16h bv = {};
    if (ncol < DK_) {
      const int kb = (lane < 16) ? 0 : 8;
#pragma unroll
      for (int hh = 0; hh < 16; ++hh)
        bv[hh] = lds_kv[(hh < 8) ? (kb + hh) : (kb + 8 + hh)][ncol];
    }
    o0 = __builtin_amdgcn_wmma_f32_16x16x32_f16(false, ap0, false, bv,
                                                (short)0, o0, false, false);
    o1 = __builtin_amdgcn_wmma_f32_16x16x32_f16(false, ap1, false, bv,
                                                (short)0, o1, false, false);
    __syncthreads();   // protect lds_kv / lds_p before next tile overwrites
  }

  // --- epilogue: O row m, col n (= d index) -> attn[b][row][h*8+n]
  if (ncol < DK_) {
#pragma unroll
    for (int r = 0; r < 8; ++r) {
      int m = r + gsel;
      attn[(((size_t)b * S_ + (q0 + m)) * E_) + h * DK_ + ncol] =
          o0[r] / lrow0[r];
      attn[(((size_t)b * S_ + (q0 + 16 + m)) * E_) + h * DK_ + ncol] =
          o1[r] / lrow1[r];
    }
  }
}

// ---------------------------------------------------------------------------
// Kernel 3: out = attn @ w_out^T.  [8192,128]x[128,128], WMMA-tiled 16x16,
// K-loop over 128 in steps of 32, f32->f16 conversion on load.
// ---------------------------------------------------------------------------
__global__ void __launch_bounds__(32, 1)
proj_kernel(const float* __restrict__ attn,
            const float* __restrict__ w_out,
            float* __restrict__ out) {
  const int lane  = threadIdx.x;
  const int bid   = blockIdx.x;
  const int ntile = bid % (E_ / 16);
  const int rtile = bid / (E_ / 16);
  const int row0  = rtile * 16;
  const int n0    = ntile * 16;
  const int mloc  = lane & 15;

  v8f acc = {};
  for (int k0 = 0; k0 < E_; k0 += 32) {
    // A fragment: rows of attn
    v16h a;
    {
      const float* ar = attn + (size_t)(row0 + mloc) * E_ + k0;
      const int kb = (lane < 16) ? 0 : 8;
#pragma unroll
      for (int hh = 0; hh < 16; ++hh) {
        int k = (hh < 8) ? (kb + hh) : (kb + 8 + hh);
        a[hh] = (_Float16)ar[k];
      }
    }
    // B fragment: column n of out = row n of w_out (out = attn @ w_out^T)
    v16h w;
    {
      const float* wr = w_out + (size_t)(n0 + mloc) * E_ + k0;
      const int kb = (lane < 16) ? 0 : 8;
#pragma unroll
      for (int hh = 0; hh < 16; ++hh)
        w[hh] = (_Float16)wr[(hh < 8) ? (kb + hh) : (kb + 8 + hh)];
    }
    acc = __builtin_amdgcn_wmma_f32_16x16x32_f16(false, a, false, w,
                                                 (short)0, acc, false, false);
  }
#pragma unroll
  for (int r = 0; r < 8; ++r) {
    int m = r + ((lane < 16) ? 0 : 8);
    out[(size_t)(row0 + m) * E_ + (n0 + mloc)] = acc[r];
  }
}

// ---------------------------------------------------------------------------
extern "C" void kernel_launch(void* const* d_in, const int* in_sizes, int n_in,
                              void* d_out, int out_size, void* d_ws, size_t ws_size,
                              hipStream_t stream) {
  const float* x     = (const float*)d_in[0];   // [4,2048,128]
  const float* theta = (const float*)d_in[1];   // [8]
  const float* w_out = (const float*)d_in[2];   // [128,128]
  float* out = (float*)d_out;                   // [4,2048,128]

  // workspace: qf (f16, 2MB) | attn (f32, 4MB)
  _Float16* qf = (_Float16*)d_ws;
  float* attn = (float*)((char*)d_ws + (size_t)B_ * H_ * S_ * DK_ * sizeof(_Float16));

  const int nfeat = B_ * S_ * H_;
  qfeat_kernel<<<(nfeat + 255) / 256, 256, 0, stream>>>(x, theta, qf);
  attn_kernel<<<B_ * H_ * (S_ / 32), 32, 0, stream>>>(qf, attn);
  proj_kernel<<<(B_ * S_ / 16) * (E_ / 16), 32, 0, stream>>>(attn, w_out, out);
}
